// GaussKernelAttn_2723009265732
// MI455X (gfx1250) — matrix operand order repaired and checked by reading, not compiled, not run on hardware
//
#include <hip/hip_runtime.h>

// GaussKernelAttn for MI455X (gfx1250, wave32, WMMA bf16 16x16x32)
//
// Stages (q is dead in the reference -> never computed):
//   K1: kv = x @ qkv_w[768:2304].T + b   -> K [b,h,n,64] bf16, V^T [b,h,64,n] bf16
//   K2: flash attention with inline Gaussian bias, online softmax -> outb bf16
//   K3: out = outb @ proj_w.T + proj_b   -> f32
//
#define PFX    8
#define NTOK   1032            // 32*32 + 8
#define BATCH  8
#define CDIM   768
#define HEADS  12
#define HD     64
#define MROWS  (BATCH * NTOK)  // 8256

typedef __attribute__((ext_vector_type(16))) __bf16       v16bf;
typedef __attribute__((ext_vector_type(8)))  float        v8f;
typedef __attribute__((ext_vector_type(4)))  float        v4f;
typedef __attribute__((ext_vector_type(4)))  unsigned int v4u;
typedef __attribute__((ext_vector_type(4)))  int          v4i;

union BFop { v4u q[2]; v16bf v; };

#if defined(__has_builtin)
#if __has_builtin(__builtin_amdgcn_global_load_async_to_lds_b128)
#define HAVE_ASYNC_LDS 1
#endif
#if __has_builtin(__builtin_amdgcn_s_wait_asynccnt)
#define HAVE_WAIT_ASYNC 1
#endif
#if __has_builtin(__builtin_amdgcn_cvt_pk_bf16_f32)
#define HAVE_CVT_PK_BF16 1
#endif
#endif

__device__ __forceinline__ unsigned short f2bf(float f) {
  unsigned int u = __builtin_bit_cast(unsigned int, f);
  u += 0x7fffu + ((u >> 16) & 1u);           // round-to-nearest-even
  return (unsigned short)(u >> 16);
}

__device__ __forceinline__ unsigned int pk2bf(float a, float b) {
#if defined(HAVE_CVT_PK_BF16)
  return __builtin_bit_cast(unsigned int, __builtin_amdgcn_cvt_pk_bf16_f32(a, b));
#else
  return (unsigned int)f2bf(a) | ((unsigned int)f2bf(b) << 16);
#endif
}

#if defined(HAVE_ASYNC_LDS)
typedef __attribute__((address_space(1))) v4i* gv4i_p;   // global int4*
typedef __attribute__((address_space(3))) v4i* lv4i_p;   // LDS int4*
__device__ __forceinline__ void cp_b128(void* lds, const void* gptr) {
  __builtin_amdgcn_global_load_async_to_lds_b128((gv4i_p)(gptr), (lv4i_p)(lds),
                                                 0, 0);
}
__device__ __forceinline__ void async_wait0() {
#if defined(HAVE_WAIT_ASYNC)
  __builtin_amdgcn_s_wait_asynccnt(0);
#else
  asm volatile("s_wait_asynccnt 0x0" ::: "memory");
#endif
}
#endif

__device__ __forceinline__ v8f wmma_bf16(v16bf a, v16bf b, v8f c) {
  return __builtin_amdgcn_wmma_f32_16x16x32_bf16(false, a, false, b,
                                                 (short)0, c, false, false);
}

__device__ __forceinline__ float rmax16(float v) {
#pragma unroll
  for (int d = 1; d < 16; d <<= 1) v = fmaxf(v, __shfl_xor(v, d, 32));
  return v;
}
__device__ __forceinline__ float rsum16(float v) {
#pragma unroll
  for (int d = 1; d < 16; d <<= 1) v += __shfl_xor(v, d, 32);
  return v;
}

// ---------------------------------------------------------------- kernel 1
__global__ __launch_bounds__(256) void qkv_kv_kernel(
    const float* __restrict__ x, const float* __restrict__ qkv_w,
    const float* __restrict__ qkv_b,
    unsigned short* __restrict__ kb, unsigned short* __restrict__ vt) {
  __shared__ __align__(16) unsigned short Ab[128 * 40];  // 32 k + 8 pad per row
  __shared__ __align__(16) unsigned short Bb[64 * 40];
  const int t = threadIdx.x, lane = t & 31, wave = t >> 5;
  const int half = lane >> 4, l16 = lane & 15;
  const int wm = wave >> 1, wn = wave & 1;               // 4 x 2 wave grid
  const int m0 = blockIdx.x * 128, n0 = blockIdx.y * 64; // n0 in [0,1536)

  v8f acc[2][2];
#pragma unroll
  for (int a = 0; a < 2; ++a)
#pragma unroll
    for (int b = 0; b < 2; ++b)
#pragma unroll
      for (int e = 0; e < 8; ++e) acc[a][b][e] = 0.0f;

  for (int k0 = 0; k0 < CDIM; k0 += 32) {
    {  // stage A: 128 x 32 f32 -> bf16 LDS
      int row = t >> 1, part = t & 1;
      int grow = m0 + row; if (grow >= MROWS) grow = MROWS - 1;
      const v4f* src = (const v4f*)(x + (size_t)grow * CDIM + k0 + part * 16);
      union { unsigned int w[8]; v4u q[2]; } pk;
#pragma unroll
      for (int qq = 0; qq < 4; ++qq) {
        v4f f = src[qq];
        pk.w[qq * 2 + 0] = pk2bf(f[0], f[1]);
        pk.w[qq * 2 + 1] = pk2bf(f[2], f[3]);
      }
      v4u* dst = (v4u*)&Ab[row * 40 + part * 16];
      dst[0] = pk.q[0]; dst[1] = pk.q[1];
    }
    {  // stage B: 64 x 32 of qkv_w rows [768 + n0 ..)
      int row = t >> 2, part = t & 3;
      const v4f* src =
          (const v4f*)(qkv_w + (size_t)(CDIM + n0 + row) * CDIM + k0 + part * 8);
      union { unsigned int w[4]; v4u q; } pk;
#pragma unroll
      for (int qq = 0; qq < 2; ++qq) {
        v4f f = src[qq];
        pk.w[qq * 2 + 0] = pk2bf(f[0], f[1]);
        pk.w[qq * 2 + 1] = pk2bf(f[2], f[3]);
      }
      *(v4u*)&Bb[row * 40 + part * 8] = pk.q;
    }
    __syncthreads();
    BFop a0, a1, b0, b1;
    const v4u* pa0 = (const v4u*)&Ab[(wm * 32 + l16) * 40];
    const v4u* pa1 = (const v4u*)&Ab[(wm * 32 + 16 + l16) * 40];
    a0.q[0] = pa0[half];        a0.q[1] = pa0[2 + half];        // A pattern
    a1.q[0] = pa1[half];        a1.q[1] = pa1[2 + half];
    const v4u* pb0 = (const v4u*)&Bb[(wn * 32 + l16) * 40];
    const v4u* pb1 = (const v4u*)&Bb[(wn * 32 + 16 + l16) * 40];
    b0.q[0] = pb0[half];        b0.q[1] = pb0[2 + half];        // B pattern
    b1.q[0] = pb1[half];        b1.q[1] = pb1[2 + half];
    acc[0][0] = wmma_bf16(a0.v, b0.v, acc[0][0]);
    acc[0][1] = wmma_bf16(a0.v, b1.v, acc[0][1]);
    acc[1][0] = wmma_bf16(a1.v, b0.v, acc[1][0]);
    acc[1][1] = wmma_bf16(a1.v, b1.v, acc[1][1]);
    __syncthreads();
  }
#pragma unroll
  for (int tm = 0; tm < 2; ++tm)
#pragma unroll
    for (int tn = 0; tn < 2; ++tn) {
      int gcol = n0 + wn * 32 + tn * 16 + l16;   // 0..1535 (k then v)
      float bias = qkv_b[CDIM + gcol];
#pragma unroll
      for (int r = 0; r < 8; ++r) {
        int grow = m0 + wm * 32 + tm * 16 + r + 8 * half;
        if (grow < MROWS) {
          unsigned short h = f2bf(acc[tm][tn][r] + bias);
          int bi = grow / NTOK, ii = grow % NTOK;
          if (gcol < CDIM) {                     // K: [b,h,n,d]
            int hh = gcol >> 6, d = gcol & 63;
            kb[(((size_t)bi * HEADS + hh) * NTOK + ii) * HD + d] = h;
          } else {                               // V transposed: [b,h,d,n]
            int cv = gcol - CDIM, hh = cv >> 6, d = cv & 63;
            vt[(((size_t)bi * HEADS + hh) * HD + d) * NTOK + ii] = h;
          }
        }
      }
    }
}

// ---------------------------------------------------------------- kernel 2
__global__ __launch_bounds__(256) void attn_kernel(
    const unsigned short* __restrict__ kb, const unsigned short* __restrict__ vt,
    unsigned short* __restrict__ outb) {
  __shared__ __align__(16) unsigned short Kt[64 * 80];       // 64 j x 64 d (+pad)
  __shared__ __align__(16) unsigned short Vt[64 * 80];       // 64 d x 64 j (+pad)
  __shared__ __align__(16) unsigned short Pt[8 * 16 * 80];   // per-wave P scratch
  const int t = threadIdx.x, lane = t & 31, wave = t >> 5;
  const int half = lane >> 4, l16 = lane & 15;
  const int bh = blockIdx.y;
  const int i0 = blockIdx.x * 128 + wave * 16;
  const unsigned short* Kb = kb + (size_t)bh * NTOK * HD;
  const unsigned short* Vb = vt + (size_t)bh * HD * NTOK;

  // A operand (16 query rows x 64 d) pinned in registers for whole kernel
  int irow = i0 + l16; if (irow >= NTOK) irow = NTOK - 1;
  const v4u* kr = (const v4u*)(Kb + (size_t)irow * HD);
  BFop aK0, aK1;
  aK0.q[0] = kr[half];     aK0.q[1] = kr[2 + half];   // d 0..31
  aK1.q[0] = kr[4 + half]; aK1.q[1] = kr[6 + half];   // d 32..63

  v8f o[4];
#pragma unroll
  for (int dt = 0; dt < 4; ++dt)
#pragma unroll
    for (int e = 0; e < 8; ++e) o[dt][e] = 0.0f;
  float mrow[8], lrow[8];
#pragma unroll
  for (int r = 0; r < 8; ++r) { mrow[r] = -3.0e38f; lrow[r] = 0.0f; }

  const int srow = t >> 2, spart = t & 3;                // staging coords

  for (int jb = 0; jb < NTOK; jb += 64) {
    {  // cooperative stage of K block and V^T block
      int jrow = jb + srow; if (jrow >= NTOK) jrow = NTOK - 1;
      const unsigned short* gk = Kb + (size_t)jrow * HD + spart * 16;
      const unsigned short* gv = Vb + (size_t)srow * NTOK + jb + spart * 16;
      unsigned short* lk = &Kt[srow * 80 + spart * 16];
      unsigned short* lv = &Vt[srow * 80 + spart * 16];
#if defined(HAVE_ASYNC_LDS)
      cp_b128(lk, gk);  cp_b128(lk + 8, gk + 8);   // ASYNCcnt path
      cp_b128(lv, gv);  cp_b128(lv + 8, gv + 8);
#else
      ((v4u*)lk)[0] = ((const v4u*)gk)[0]; ((v4u*)lk)[1] = ((const v4u*)gk)[1];
      ((v4u*)lv)[0] = ((const v4u*)gv)[0]; ((v4u*)lv)[1] = ((const v4u*)gv)[1];
#endif
      // prefetch next block while this one is consumed
      if (jb + 64 < NTOK) {
        __builtin_prefetch(Kb + (size_t)(jb + 64 + srow) * HD, 0, 3);
        __builtin_prefetch(Vb + (size_t)srow * NTOK + jb + 64, 0, 3);
      }
    }
#if defined(HAVE_ASYNC_LDS)
    async_wait0();
#endif
    __syncthreads();

    // scores S[16 i x 64 j] = Ki . Kj  (8 WMMAs)
    v8f s4[4];
#pragma unroll
    for (int jt = 0; jt < 4; ++jt) {
      const v4u* kj = (const v4u*)&Kt[(jt * 16 + l16) * 80];
      BFop b0, b1;
      b0.q[0] = kj[half];         b0.q[1] = kj[2 + half];          // d 0..31
      b1.q[0] = kj[4 + half];     b1.q[1] = kj[6 + half];          // d 32..63
      v8f z;
#pragma unroll
      for (int e = 0; e < 8; ++e) z[e] = 0.0f;
      z = wmma_bf16(aK0.v, b0.v, z);
      z = wmma_bf16(aK1.v, b1.v, z);
      s4[jt] = z;
    }
    // scale + Gaussian bias + tail mask
#pragma unroll
    for (int jt = 0; jt < 4; ++jt)
#pragma unroll
      for (int r = 0; r < 8; ++r) {
        int ig = i0 + r + 8 * half;
        int jg = jb + jt * 16 + l16;
        float val = s4[jt][r] * 0.125f;
        if (ig >= PFX && jg >= PFX) {
          int ri = (ig - PFX) >> 5, ci = (ig - PFX) & 31;
          int rj = (jg - PFX) >> 5, cj = (jg - PFX) & 31;
          int dr = ri - rj, dc = ci - cj;
          val += __expf(-0.02f * (float)(dr * dr + dc * dc));
        }
        if (jg >= NTOK) val = -3.0e38f;
        s4[jt][r] = val;
      }
    // online softmax (rows live in 16-lane halves -> shfl_xor reductions)
    float alpha[8], rs[8];
#pragma unroll
    for (int r = 0; r < 8; ++r) {
      float bm = fmaxf(fmaxf(s4[0][r], s4[1][r]), fmaxf(s4[2][r], s4[3][r]));
      bm = rmax16(bm);
      float mn = fmaxf(mrow[r], bm);
      alpha[r] = __expf(mrow[r] - mn);
      mrow[r] = mn;
      rs[r] = 0.0f;
    }
#pragma unroll
    for (int jt = 0; jt < 4; ++jt)
#pragma unroll
      for (int r = 0; r < 8; ++r) {
        float p = __expf(s4[jt][r] - mrow[r]);
        s4[jt][r] = p;
        rs[r] += p;
      }
#pragma unroll
    for (int r = 0; r < 8; ++r) {
      lrow[r] = lrow[r] * alpha[r] + rsum16(rs[r]);
#pragma unroll
      for (int dt = 0; dt < 4; ++dt) o[dt][r] *= alpha[r];
    }
    // P: C-layout -> A-layout via per-wave LDS scratch (wave-local dscnt wait)
    unsigned short* Pr = &Pt[wave * 16 * 80];
#pragma unroll
    for (int jt = 0; jt < 4; ++jt)
#pragma unroll
      for (int r = 0; r < 8; ++r)
        Pr[(r + 8 * half) * 80 + jt * 16 + l16] = f2bf(s4[jt][r]);
    asm volatile("s_wait_dscnt 0x0" ::: "memory");
    BFop aP0, aP1;
    const v4u* pp = (const v4u*)&Pr[l16 * 80];
    aP0.q[0] = pp[half];     aP0.q[1] = pp[2 + half];   // j 0..31
    aP1.q[0] = pp[4 + half]; aP1.q[1] = pp[6 + half];   // j 32..63
    // O += P @ V (8 WMMAs), B operand from V^T rows
#pragma unroll
    for (int dt = 0; dt < 4; ++dt) {
      const v4u* vv = (const v4u*)&Vt[(dt * 16 + l16) * 80];
      BFop b0, b1;
      b0.q[0] = vv[half];         b0.q[1] = vv[2 + half];
      b1.q[0] = vv[4 + half];     b1.q[1] = vv[6 + half];
      o[dt] = wmma_bf16(aP0.v, b0.v, o[dt]);
      o[dt] = wmma_bf16(aP1.v, b1.v, o[dt]);
    }
    __syncthreads();
  }
  // epilogue: normalize and store bf16 [b, n, h*64+d]
  const int bi = bh / HEADS, hh = bh % HEADS;
#pragma unroll
  for (int dt = 0; dt < 4; ++dt)
#pragma unroll
    for (int r = 0; r < 8; ++r) {
      int ig = i0 + r + 8 * half;
      if (ig < NTOK) {
        float v = o[dt][r] / lrow[r];
        outb[((size_t)bi * NTOK + ig) * CDIM + hh * HD + dt * 16 + l16] = f2bf(v);
      }
    }
}

// ---------------------------------------------------------------- kernel 3
__global__ __launch_bounds__(256) void proj_kernel(
    const unsigned short* __restrict__ ob, const float* __restrict__ pw,
    const float* __restrict__ pb, float* __restrict__ out) {
  __shared__ __align__(16) unsigned short Ab[128 * 40];
  __shared__ __align__(16) unsigned short Bb[64 * 40];
  const int t = threadIdx.x, lane = t & 31, wave = t >> 5;
  const int half = lane >> 4, l16 = lane & 15;
  const int wm = wave >> 1, wn = wave & 1;
  const int m0 = blockIdx.x * 128, n0 = blockIdx.y * 64;

  v8f acc[2][2];
#pragma unroll
  for (int a = 0; a < 2; ++a)
#pragma unroll
    for (int b = 0; b < 2; ++b)
#pragma unroll
      for (int e = 0; e < 8; ++e) acc[a][b][e] = 0.0f;

  for (int k0 = 0; k0 < CDIM; k0 += 32) {
    {  // stage A (already bf16): async byte copy into LDS
      int row = t >> 1, part = t & 1;
      int grow = m0 + row; if (grow >= MROWS) grow = MROWS - 1;
      const unsigned short* ga = ob + (size_t)grow * CDIM + k0 + part * 16;
      unsigned short* la = &Ab[row * 40 + part * 16];
#if defined(HAVE_ASYNC_LDS)
      cp_b128(la, ga);  cp_b128(la + 8, ga + 8);
#else
      ((v4u*)la)[0] = ((const v4u*)ga)[0]; ((v4u*)la)[1] = ((const v4u*)ga)[1];
#endif
    }
    {  // stage B from proj_w (f32 -> bf16)
      int row = t >> 2, part = t & 3;
      const v4f* src = (const v4f*)(pw + (size_t)(n0 + row) * CDIM + k0 + part * 8);
      union { unsigned int w[4]; v4u q; } pk;
#pragma unroll
      for (int qq = 0; qq < 2; ++qq) {
        v4f f = src[qq];
        pk.w[qq * 2 + 0] = pk2bf(f[0], f[1]);
        pk.w[qq * 2 + 1] = pk2bf(f[2], f[3]);
      }
      *(v4u*)&Bb[row * 40 + part * 8] = pk.q;
    }
#if defined(HAVE_ASYNC_LDS)
    async_wait0();
#endif
    __syncthreads();
    BFop a0, a1, b0, b1;
    const v4u* pa0 = (const v4u*)&Ab[(wm * 32 + l16) * 40];
    const v4u* pa1 = (const v4u*)&Ab[(wm * 32 + 16 + l16) * 40];
    a0.q[0] = pa0[half];        a0.q[1] = pa0[2 + half];
    a1.q[0] = pa1[half];        a1.q[1] = pa1[2 + half];
    const v4u* pb0 = (const v4u*)&Bb[(wn * 32 + l16) * 40];
    const v4u* pb1 = (const v4u*)&Bb[(wn * 32 + 16 + l16) * 40];
    b0.q[0] = pb0[half];        b0.q[1] = pb0[2 + half];
    b1.q[0] = pb1[half];        b1.q[1] = pb1[2 + half];
    acc[0][0] = wmma_bf16(a0.v, b0.v, acc[0][0]);
    acc[0][1] = wmma_bf16(a0.v, b1.v, acc[0][1]);
    acc[1][0] = wmma_bf16(a1.v, b0.v, acc[1][0]);
    acc[1][1] = wmma_bf16(a1.v, b1.v, acc[1][1]);
    __syncthreads();
  }
#pragma unroll
  for (int tm = 0; tm < 2; ++tm)
#pragma unroll
    for (int tn = 0; tn < 2; ++tn) {
      int gcol = n0 + wn * 32 + tn * 16 + l16;
      float bias = pb[gcol];
#pragma unroll
      for (int r = 0; r < 8; ++r) {
        int grow = m0 + wm * 32 + tm * 16 + r + 8 * half;
        if (grow < MROWS)
          out[(size_t)grow * CDIM + gcol] = acc[tm][tn][r] + bias;
      }
    }
}

// ---------------------------------------------------------------- launch
extern "C" void kernel_launch(void* const* d_in, const int* in_sizes, int n_in,
                              void* d_out, int out_size, void* d_ws, size_t ws_size,
                              hipStream_t stream) {
  const float* x      = (const float*)d_in[0];
  const float* qkv_w  = (const float*)d_in[1];
  const float* qkv_b  = (const float*)d_in[2];
  const float* proj_w = (const float*)d_in[3];
  const float* proj_b = (const float*)d_in[4];

  const size_t slab = (size_t)BATCH * HEADS * NTOK * HD;  // 6,340,608 elems
  unsigned short* kb = (unsigned short*)d_ws;             // K  [b,h,n,d]
  unsigned short* vt = kb + slab;                         // V^T[b,h,d,n]
  unsigned short* ob = vt + slab;                         // attn out bf16 (same size)

  qkv_kv_kernel<<<dim3((MROWS + 127) / 128, (2 * CDIM) / 64), 256, 0, stream>>>(
      x, qkv_w, qkv_b, kb, vt);
  attn_kernel<<<dim3((NTOK + 127) / 128, BATCH * HEADS), 256, 0, stream>>>(
      kb, vt, ob);
  proj_kernel<<<dim3((MROWS + 127) / 128, CDIM / 64), 256, 0, stream>>>(
      ob, proj_w, proj_b, (float*)d_out);
}
